// IntraOptionPolicy_4947802325037
// MI455X (gfx1250) — matrix operand order repaired and checked by reading, not compiled, not run on hardware
//
#include <hip/hip_runtime.h>
#include <hip/hip_bf16.h>

#define O_ 1024
#define D_ 1024
#define E_ 16
#define K_ 4
#define H_ 2048
#define A_ 64
#define CAP 1024
#define TILE_M 64
#define TILES (CAP / TILE_M)   // 16 tiles per expert

typedef __attribute__((ext_vector_type(16))) _Float16 v16h;
typedef __attribute__((ext_vector_type(8)))  float    v8f;

// ---------------------------------------------------------------------------
// CDNA5 async global->LDS copy (ASYNCcnt path, CDNA5 ISA)
// LDS address operand = addr[31:0] of the generic pointer (flat aperture rule).
// ---------------------------------------------------------------------------
__device__ inline void async_copy_b128(void* lds_dst, const void* gsrc) {
  unsigned ldsoff = (unsigned)(size_t)lds_dst;
  asm volatile("global_load_async_to_lds_b128 %0, %1, off"
               :: "v"(ldsoff), "v"(gsrc)
               : "memory");
}
__device__ inline void wait_async0() {
  asm volatile("s_wait_asynccnt 0x0" ::: "memory");
}

// ---------------------------------------------------------------------------
// WMMA helpers (layouts per CDNA5 ISA §7.12.2, wave32)
// ---------------------------------------------------------------------------
__device__ inline v8f wmma_f16(v16h a, v16h b, v8f c) {
  // D = A(16x32 f16) * B(32x16 f16) + C(16x16 f32)
  return __builtin_amdgcn_wmma_f32_16x16x32_f16(
      /*neg_a=*/false, a, /*neg_b=*/false, b,
      /*c_mod=*/(short)0, c, /*reuse_a=*/false, /*reuse_b=*/false);
}

// A fragment: lane l holds row m=l&15; lanes<16 carry K {0..7,16..23},
// lanes>=16 carry K {8..15,24..31}. 'row' points at the 32-wide K chunk.
__device__ inline v16h load_a_lds(const _Float16* row, int koff) {
  v16h r;
#pragma unroll
  for (int i = 0; i < 8; ++i) {
    r[i]     = row[koff + i];
    r[i + 8] = row[koff + 16 + i];
  }
  return r;
}

// B fragment (32x16): lane l holds N=l&15; lanes<16 carry K {0..7,16..23},
// lanes>=16 carry K {8..15,24..31} (row stride ld).
// Source is f32 in global memory; convert to f16 on load (RTE).
__device__ inline v16h load_b_f32(const float* __restrict__ p, int ld) {
  v16h r;
#pragma unroll
  for (int i = 0; i < 8; ++i) {
    r[i]     = (_Float16)p[(size_t)i * ld];
    r[i + 8] = (_Float16)p[(size_t)(16 + i) * ld];
  }
  return r;
}

// ---------------------------------------------------------------------------
// Kernel 1: gating — logits = obs[o]·w_gate[o], top-4 softmax, expert lists
// ---------------------------------------------------------------------------
__global__ __launch_bounds__(256) void gating_kernel(
    const float* __restrict__ obs, const float* __restrict__ w_gate,
    int* __restrict__ counts, int* __restrict__ eopt,
    float* __restrict__ egate, int* __restrict__ epk,
    float* __restrict__ importance, float* __restrict__ loadv) {
  int o = blockIdx.x;
  int t = threadIdx.x;

  __shared__ __align__(16) float xs[D_];
  const float* x = obs + (size_t)o * D_;
  // 1024 f32 = 256 x b128 async transfers, one per thread
  async_copy_b128(&xs[t * 4], x + t * 4);
  wait_async0();
  __syncthreads();

  // w_gate[o] is [D][E] row-major: idx = d*16 + e; thread t owns e = t&15.
  const float* wg = w_gate + (size_t)o * (D_ * E_);
  float acc = 0.f;
#pragma unroll 4
  for (int i = 0; i < (D_ * E_) / 256; ++i) {
    int idx = t + i * 256;                 // fully coalesced stream
    acc += wg[idx] * xs[idx >> 4];
  }

  __shared__ float part[256];
  part[t] = acc;
  __syncthreads();
  for (int s = 128; s >= 16; s >>= 1) {    // strides are multiples of 16:
    if (t < s) part[t] += part[t + s];     // expert lane stays aligned
    __syncthreads();
  }

  if (t == 0) {
    float lg[16];
#pragma unroll
    for (int e = 0; e < 16; ++e) lg[e] = part[e];
    int ti[4]; float tv[4]; unsigned mask = 0;
#pragma unroll
    for (int k = 0; k < 4; ++k) {          // stable top-4 (ties -> low index)
      float best = -3.4e38f; int bi = 0;
      for (int e = 0; e < 16; ++e)
        if (!((mask >> e) & 1u) && lg[e] > best) { best = lg[e]; bi = e; }
      ti[k] = bi; tv[k] = best; mask |= 1u << bi;
    }
    float mx = tv[0], ssum = 0.f, ex[4];
#pragma unroll
    for (int k = 0; k < 4; ++k) { ex[k] = expf(tv[k] - mx); ssum += ex[k]; }
    float impv = 0.f, ldc = 0.f;
#pragma unroll
    for (int k = 0; k < 4; ++k) {
      float g = ex[k] / ssum;
      impv += g;
      ldc += (g > 0.f) ? 1.f : 0.f;
      int slot = atomicAdd(&counts[ti[k]], 1);
      int base = ti[k] * CAP + slot;
      eopt[base]  = o;
      egate[base] = g;
      epk[base]   = o * K_ + k;            // deterministic output slot
    }
    importance[o] = impv;
    loadv[o]      = ldc;
  }
}

// ---------------------------------------------------------------------------
// Kernel 2: loss = 0.01*(cv^2(importance) + cv^2(load)), ddof=1
// ---------------------------------------------------------------------------
__global__ __launch_bounds__(256) void loss_kernel(
    const float* __restrict__ imp, const float* __restrict__ ld,
    float* __restrict__ loss) {
  int t = threadIdx.x;
  float s1 = 0, s2 = 0, s3 = 0, s4 = 0;
  for (int i = t; i < O_; i += 256) {
    float a = imp[i]; s1 += a; s2 += a * a;
    float b = ld[i];  s3 += b; s4 += b * b;
  }
  __shared__ float r1[256], r2[256], r3[256], r4[256];
  r1[t] = s1; r2[t] = s2; r3[t] = s3; r4[t] = s4;
  __syncthreads();
  for (int s = 128; s >= 1; s >>= 1) {
    if (t < s) { r1[t] += r1[t+s]; r2[t] += r2[t+s]; r3[t] += r3[t+s]; r4[t] += r4[t+s]; }
    __syncthreads();
  }
  if (t == 0) {
    const float n = (float)O_;
    float m1 = r1[0] / n, v1 = (r2[0] - r1[0] * r1[0] / n) / (n - 1.f);
    float m2 = r3[0] / n, v2 = (r4[0] - r3[0] * r3[0] / n) / (n - 1.f);
    loss[0] = 0.01f * (v1 / (m1 * m1 + 1e-10f) + v2 / (m2 * m2 + 1e-10f));
  }
}

// ---------------------------------------------------------------------------
// Kernel 3: fused expert MLP via WMMA.
// Block = (expert e, 64-row tile of e's gathered options), 4 waves x 16 rows.
// y_part[(o*4+k)] = gate * (relu(x·W1+b1)·W2 + b2)
// ---------------------------------------------------------------------------
__global__ __launch_bounds__(128) void moe_kernel(
    const float* __restrict__ obs, const float* __restrict__ w1,
    const float* __restrict__ b1, const float* __restrict__ w2,
    const float* __restrict__ b2, const int* __restrict__ counts,
    const int* __restrict__ eopt, const float* __restrict__ egate,
    const int* __restrict__ epk, float* __restrict__ y_part) {
  int e = blockIdx.x / TILES;
  int tile = blockIdx.x % TILES;
  int nValid = counts[e];
  int rowStart = tile * TILE_M;
  if (rowStart >= nValid) return;

  extern __shared__ _Float16 lds[];
  _Float16* Xh = lds;                          // [64][1024] f16 X tile, 128KB
  _Float16* Hh = lds + TILE_M * D_;            // [4 waves][16][32] relu bounce, 4KB
  float* Xstage = (float*)(lds + TILE_M * D_ + 4 * 16 * 32); // [4][1024] f32, 16KB

  __shared__ int   so[TILE_M];
  __shared__ float sgate[TILE_M];
  __shared__ int   spk[TILE_M];

  int t = threadIdx.x;
  if (t < TILE_M) {
    int idx = rowStart + t;
    bool v = idx < nValid;
    int base = e * CAP + idx;
    so[t]    = v ? eopt[base] : 0;
    sgate[t] = v ? egate[base] : 0.f;
    spk[t]   = v ? epk[base] : 0;
  }
  __syncthreads();

  // Stage X: async-copy 4 gathered f32 rows into LDS, convert to resident f16.
  for (int rb = 0; rb < TILE_M; rb += 4) {
    // 4 rows * 1024 f32 = 1024 b128 transfers; 128 threads -> 8 each
    for (int j = t; j < 1024; j += 128) {
      int r  = j >> 8;              // 256 b128 chunks per row
      int c4 = (j & 255) * 4;
      async_copy_b128(&Xstage[r * D_ + c4],
                      obs + (size_t)so[rb + r] * D_ + c4);
    }
    wait_async0();
    __syncthreads();
    for (int j = t; j < 4 * D_; j += 128) {
      Xh[(size_t)(rb + (j >> 10)) * D_ + (j & (D_ - 1))] = (_Float16)Xstage[j];
    }
    __syncthreads();
  }

  int wv = t >> 5, lane = t & 31;
  int mrow  = wv * 16;
  int n     = lane & 15;                   // C-fragment N / A-fragment row
  int khalf = (lane < 16) ? 0 : 8;         // A-fragment K half select
  int mbase = (lane < 16) ? 0 : 8;         // C-fragment M base

  const float* w1e = w1 + (size_t)e * D_ * H_;
  const float* w2e = w2 + (size_t)e * H_ * A_;
  const float* b1e = b1 + e * H_;
  _Float16* hrow = Hh + wv * (16 * 32);

  v8f y0 = {}, y1 = {}, y2 = {}, y3 = {};

  for (int hb = 0; hb < H_; hb += 32) {
    // ---- GEMM1: Htile[16x32] = X[16x1024] * W1[:, hb:hb+32] ----
    v8f c0 = {}, c1 = {};
    for (int dc = 0; dc < D_; dc += 32) {
      v16h a = load_a_lds(Xh + (size_t)(mrow + n) * D_ + dc, khalf);
      const float* bp = w1e + (size_t)(dc + khalf) * H_ + hb + n;  // n = lane&15
      c0 = wmma_f16(a, load_b_f32(bp, H_),      c0);
      c1 = wmma_f16(a, load_b_f32(bp + 16, H_), c1);
    }
    // bias + relu, bounce through LDS to re-lay as A-fragment of GEMM2
    float bias0 = b1e[hb + n];
    float bias1 = b1e[hb + 16 + n];
#pragma unroll
    for (int r = 0; r < 8; ++r) {
      int m = r + mbase;
      float v0 = c0[r] + bias0; v0 = v0 > 0.f ? v0 : 0.f;
      float v1 = c1[r] + bias1; v1 = v1 > 0.f ? v1 : 0.f;
      hrow[m * 32 + n]      = (_Float16)v0;
      hrow[m * 32 + 16 + n] = (_Float16)v1;
    }
    __syncthreads();                       // order LDS RAW safely
    // ---- GEMM2: y += Htile[16x32] * W2[hb:hb+32, 0:64] ----
    v16h a2 = load_a_lds(hrow + n * 32, khalf);
    const float* w2row = w2e + (size_t)(hb + khalf) * A_ + n; // n = lane&15
    y0 = wmma_f16(a2, load_b_f32(w2row +  0, A_), y0);
    y1 = wmma_f16(a2, load_b_f32(w2row + 16, A_), y1);
    y2 = wmma_f16(a2, load_b_f32(w2row + 32, A_), y2);
    y3 = wmma_f16(a2, load_b_f32(w2row + 48, A_), y3);
    __syncthreads();
  }

  const float* b2e = b2 + e * A_;
#pragma unroll
  for (int r = 0; r < 8; ++r) {
    int rt = mrow + r + mbase;
    if (rowStart + rt < nValid) {
      float g = sgate[rt];
      float* dst = y_part + (size_t)spk[rt] * A_;
      dst[ 0 + n] = g * (y0[r] + b2e[ 0 + n]);
      dst[16 + n] = g * (y1[r] + b2e[16 + n]);
      dst[32 + n] = g * (y2[r] + b2e[32 + n]);
      dst[48 + n] = g * (y3[r] + b2e[48 + n]);
    }
  }
}

// ---------------------------------------------------------------------------
// Kernel 4: deterministic combine y = sum_k y_part[o,k]
// ---------------------------------------------------------------------------
__global__ __launch_bounds__(256) void combine_kernel(
    const float* __restrict__ yp, float* __restrict__ y) {
  int i = blockIdx.x * 256 + threadIdx.x;  // i < O_*A_
  int o = i >> 6, a = i & 63;
  float s = 0.f;
#pragma unroll
  for (int k = 0; k < K_; ++k) s += yp[((size_t)(o * K_ + k)) * A_ + a];
  y[i] = s;
}

// ---------------------------------------------------------------------------
// Kernel 5: action head — mu = obs[o]·last_w[o] + last_b, softmax + log
// ---------------------------------------------------------------------------
__global__ __launch_bounds__(256) void action_kernel(
    const float* __restrict__ obs, const float* __restrict__ lw,
    const float* __restrict__ lb, float* __restrict__ probs,
    float* __restrict__ logp) {
  int o = blockIdx.x;
  int t = threadIdx.x;
  int a = t & 63, dg = t >> 6;             // 4 d-groups x 64 columns
  const float* W = lw + (size_t)o * D_ * A_;
  const float* x = obs + (size_t)o * D_;
  float acc = 0.f;
  for (int d = dg; d < D_; d += 4) {
    __builtin_prefetch(&W[(size_t)(d + 32) * A_ + a], 0, 1); // global_prefetch_b8
    acc += x[d] * W[(size_t)d * A_ + a];
  }
  __shared__ float sh[256];
  sh[t] = acc;
  __syncthreads();
  if (t < 128) sh[t] += sh[t + 128];
  __syncthreads();
  __shared__ float mu[64];
  if (t < 64) mu[t] = sh[t] + sh[t + 64] + lb[(size_t)o * A_ + t];
  __syncthreads();
  // softmax over 64
  if (t < 64) sh[t] = mu[t];
  __syncthreads();
  for (int s = 32; s >= 1; s >>= 1) {
    if (t < s) sh[t] = fmaxf(sh[t], sh[t + s]);
    __syncthreads();
  }
  float mx = sh[0];
  __syncthreads();
  __shared__ float es[64];
  if (t < 64) { es[t] = expf(mu[t] - mx); sh[t] = es[t]; }
  __syncthreads();
  for (int s = 32; s >= 1; s >>= 1) {
    if (t < s) sh[t] += sh[t + s];
    __syncthreads();
  }
  float denom = sh[0];
  if (t < 64) {
    float p = es[t] / denom;
    probs[(size_t)o * A_ + t] = p;
    float zadd = (p == 0.0f) ? 1e-8f : 0.0f;
    logp[(size_t)o * A_ + t] = logf(p + zadd);
  }
}

// ---------------------------------------------------------------------------
extern "C" void kernel_launch(void* const* d_in, const int* in_sizes, int n_in,
                              void* d_out, int out_size, void* d_ws, size_t ws_size,
                              hipStream_t stream) {
  const float* obs   = (const float*)d_in[0];
  const float* wgate = (const float*)d_in[1];
  const float* w1    = (const float*)d_in[2];
  const float* b1    = (const float*)d_in[3];
  const float* w2    = (const float*)d_in[4];
  const float* b2    = (const float*)d_in[5];
  const float* lw    = (const float*)d_in[6];
  const float* lb    = (const float*)d_in[7];

  float* out   = (float*)d_out;
  float* probs = out;
  float* logp  = out + O_ * A_;
  float* y     = out + 2 * O_ * A_;
  float* loss  = out + 3 * O_ * A_;

  char* ws = (char*)d_ws;
  int*   counts     = (int*)ws;                                   // 16 ints
  float* importance = (float*)(ws + 256);                         // O floats
  float* loadv      = (float*)(ws + 256 + 4096);                  // O floats
  int*   eopt       = (int*)(ws + 256 + 8192);                    // E*CAP
  float* egate      = (float*)(ws + 256 + 8192 + 65536);          // E*CAP
  int*   epk        = (int*)(ws + 256 + 8192 + 2 * 65536);        // E*CAP
  float* ypart      = (float*)(ws + 256 + 8192 + 3 * 65536);      // O*K*A

  hipMemsetAsync(counts, 0, E_ * sizeof(int), stream);

  gating_kernel<<<O_, 256, 0, stream>>>(obs, wgate, counts, eopt, egate, epk,
                                        importance, loadv);
  loss_kernel<<<1, 256, 0, stream>>>(importance, loadv, loss);

  size_t ldsz = (size_t)(TILE_M * D_ + 4 * 16 * 32) * sizeof(_Float16)
              + (size_t)4 * D_ * sizeof(float);                   // ~148KB
  hipFuncSetAttribute(reinterpret_cast<const void*>(moe_kernel),
                      hipFuncAttributeMaxDynamicSharedMemorySize, (int)ldsz);
  moe_kernel<<<E_ * TILES, 128, ldsz, stream>>>(obs, w1, b1, w2, b2, counts,
                                                eopt, egate, epk, ypart);
  combine_kernel<<<(O_ * A_) / 256, 256, 0, stream>>>(ypart, y);
  action_kernel<<<O_, 256, 0, stream>>>(obs, lw, lb, probs, logp);
}
